// EmotionCausalModel_8658654069217
// MI455X (gfx1250) — matrix operand order repaired and checked by reading, not compiled, not run on hardware
//
#include <hip/hip_runtime.h>
#include <stdint.h>

typedef __attribute__((ext_vector_type(8)))  _Float16 v8h;
typedef __attribute__((ext_vector_type(16))) _Float16 v16h;
typedef __attribute__((ext_vector_type(8)))  float    v8f;

// ---------------- model dimensions ----------------
constexpr int CB    = 8;            // batch
constexpr int CONV  = 24;           // conversation length
constexpr int UTLEN = 64;           // utterance length (T)
constexpr int WEMB  = 512;          // word embed
constexpr int SEMB  = 64;           // speaker embed
constexpr int EEMB  = 64;           // emotion embed
constexpr int UTE   = 512;          // utterance LSTM hidden
constexpr int SHH   = 256;          // span LSTM hidden
constexpr int NSEQ  = CB * CONV;    // 192
constexpr int NARCS = 2400;
constexpr int UTCAT = 2 * UTE + SEMB;   // 1088 (K for FFN, /32 ok)
constexpr int BIA   = 544;              // biaffine K/N padded (513 -> 544)
constexpr int BKT   = BIA / 32;         // 17
constexpr int BNT   = BIA / 16;         // 34
constexpr int SEM_ELEMS = CB * CONV * CONV * 7;  // 32256

#define DEVI __device__ __forceinline__

// ---------------- WMMA fragment helpers (CDNA5 16x16x32 f16, wave32) --------
DEVI v16h make16(v8h lo, v8h hi) {
  return __builtin_shufflevector(lo, hi, 0,1,2,3,4,5,6,7,8,9,10,11,12,13,14,15);
}

// A (16x32 MxK) fragment, row-major source; rowp points at this lane's row
// (row = lane&15).  layout: elem e -> k = kbase + (lane/16)*16 + e
DEVI v16h a_frag(const _Float16* rowp, int kbase, int lane) {
  const _Float16* p = rowp + kbase + ((lane >> 4) << 4);
  v8h lo = *(const v8h*)p;
  v8h hi = *(const v8h*)(p + 8);
  return make16(lo, hi);
}

// B (32x16 KxN) fragment from pre-packed buffer: 16 contiguous halves per lane.
DEVI v16h b_frag_packed(const _Float16* pack, int f, int lane) {
  const _Float16* p = pack + (((size_t)f << 5) + (size_t)lane) * 16;
  v8h lo = *(const v8h*)p;
  v8h hi = *(const v8h*)(p + 8);
  return make16(lo, hi);
}

// B fragment where B[k][n] = M[n][k], M row-major; rowp = lane's row (n = lane&15).
DEVI v16h b_frag_rowT(const _Float16* rowp, int kbase, int lane) {
  const _Float16* p = rowp + kbase + ((lane >> 4) << 4);
  v8h lo = *(const v8h*)p;
  v8h hi = *(const v8h*)(p + 8);
  return make16(lo, hi);
}

DEVI v8f wmma_f16(v16h a, v16h b, v8f c) {
  return __builtin_amdgcn_wmma_f32_16x16x32_f16(false, a, false, b, (short)0, c,
                                                false, false);
}

DEVI float sigf(float x) { return 1.f / (1.f + __expf(-x)); }

// LDS byte offset of a generic pointer to a __shared__ object (addrspace(3) cast)
DEVI uint32_t lds_off(const void* p) {
  return (uint32_t)(uintptr_t)(const __attribute__((address_space(3))) void*)p;
}

// issue one 64B async global->LDS copy (4 x B128), tracked by ASYNCcnt
DEVI void async_copy64(uint32_t lds, const _Float16* gsrc) {
  unsigned long long ga = (unsigned long long)(uintptr_t)gsrc;
  asm volatile(
      "global_load_async_to_lds_b128 %0, %1, off\n\t"
      "global_load_async_to_lds_b128 %0, %1, off offset:16\n\t"
      "global_load_async_to_lds_b128 %0, %1, off offset:32\n\t"
      "global_load_async_to_lds_b128 %0, %1, off offset:48"
      :: "v"(lds), "v"(ga) : "memory");
}

// ---------------- prep kernels ----------------------------------------------
__global__ __launch_bounds__(256) void embed_mean_kernel(
    const int* __restrict__ words, const float* __restrict__ table,
    _Float16* __restrict__ out) {
  int idx = blockIdx.x * 256 + threadIdx.x;
  if (idx >= NSEQ * UTLEN * WEMB) return;
  int c  = idx & (WEMB - 1);
  int st = idx >> 9;                       // seq*64 + t
  const int* w4 = words + (size_t)st * 4;
  float s = 0.f;
#pragma unroll
  for (int q = 0; q < 4; ++q) s += table[(size_t)w4[q] * WEMB + c];
  out[idx] = (_Float16)(0.25f * s);
}

// pack fp32 W[N][K] (used as B = W^T) into WMMA B-fragment layout, f16
__global__ __launch_bounds__(256) void pack_weight_kernel(
    const float* __restrict__ W, _Float16* __restrict__ out, int N, int K) {
  int idx = blockIdx.x * 256 + threadIdx.x;
  if (idx >= N * K) return;
  int e = idx & 15, lane = (idx >> 4) & 31, frag = idx >> 9;
  int KT = K >> 5;
  int nt = frag / KT, kt = frag - nt * KT;
  int n = nt * 16 + (lane & 15);
  int k = kt * 32 + ((lane >> 4) << 4) + e;
  out[idx] = (_Float16)W[(size_t)n * K + k];
}

// pack Wbi [7][513][513] -> [7][34 nt][17 kt][32][16] f16, zero padded
__global__ __launch_bounds__(256) void pack_wbi_kernel(
    const float* __restrict__ Wbi, _Float16* __restrict__ out) {
  int idx = blockIdx.x * 256 + threadIdx.x;
  if (idx >= 7 * BIA * BIA) return;
  int e = idx & 15, lane = (idx >> 4) & 31;
  int frag = (idx >> 9) % (BNT * BKT);
  int o = idx / (BIA * BIA);
  int nt = frag / BKT, kt = frag - nt * BKT;
  int j = nt * 16 + (lane & 15);                 // Wbi second index (N)
  int i = kt * 32 + ((lane >> 4) << 4) + e;      // Wbi first index (K)
  float v = (i < 513 && j < 513) ? Wbi[((size_t)o * 513 + i) * 513 + j] : 0.f;
  out[idx] = (_Float16)v;
}

__global__ __launch_bounds__(256) void spk_fill_kernel(
    const int* __restrict__ speakers, const float* __restrict__ spk_table,
    _Float16* __restrict__ ut) {
  int idx = blockIdx.x * 256 + threadIdx.x;
  if (idx >= NSEQ * SEMB) return;
  int seq = idx >> 6, c = idx & 63;
  ut[(size_t)seq * UTCAT + 2 * UTE + c] =
      (_Float16)spk_table[(size_t)speakers[seq] * SEMB + c];
}

// eff1/cau1 padding: col 512 = 1.0 (biaffine bias), cols 513..543 = 0
__global__ __launch_bounds__(256) void pad_fill_kernel(_Float16* __restrict__ e1,
                                                       _Float16* __restrict__ c1) {
  int idx = blockIdx.x * 256 + threadIdx.x;
  if (idx >= NSEQ * 32) return;
  int row = idx >> 5, c = idx & 31;
  _Float16 v = (_Float16)((c == 0) ? 1.f : 0.f);
  e1[(size_t)row * BIA + 512 + c] = v;
  c1[(size_t)row * BIA + 512 + c] = v;
}

// ---------------- fused WMMA BiLSTM (both directions in one launch) ----------
// One WG = 16 sequences x one direction, 8 waves.  Each wave owns HID/8 hidden
// cols and computes all 4 gates for them -> cell update entirely in registers;
// x_t is double-buffered in LDS and prefetched with async global->LDS copies.
template <bool SPAN>
__global__ __launch_bounds__(256) void lstm_wmma_kernel(
    const _Float16* __restrict__ we,       // [192][64][512] f16 word embeds
    const _Float16* __restrict__ wihpF, const _Float16* __restrict__ whhpF,
    const float* __restrict__ biasF,
    const _Float16* __restrict__ wihpB, const _Float16* __restrict__ whhpB,
    const float* __restrict__ biasB,
    const int* __restrict__ emotions,      // SPAN only [192]
    const float* __restrict__ em_table,    // SPAN only [7][64]
    const float* __restrict__ wout_all,    // SPAN only: sWout [512]
    float* __restrict__ dotF,              // SPAN only: [2400][64]
    float* __restrict__ dotB,              // SPAN only: [2400][64]
    _Float16* __restrict__ hout,           // !SPAN: ut_f16
    int nblk_dir) {
  constexpr int INK = SPAN ? (WEMB + EEMB) : WEMB;  // 576 / 512
  constexpr int HID = SPAN ? SHH : UTE;             // 256 / 512
  constexpr int KTX = INK / 32, KTH = HID / 32;
  constexpr int NTG = HID / 16;   // n-tiles per gate (whole block)
  constexpr int CPW = HID / 8;    // hidden cols per wave
  constexpr int TPG = CPW / 16;   // tiles per gate per wave

  __shared__ _Float16 xt[2][16][INK];
  __shared__ _Float16 ht[16][HID];
  __shared__ float rowsum[16];

  const int tid = threadIdx.x;
  const int wave = tid >> 5, lane = tid & 31;
  const bool rev = (int)blockIdx.x >= nblk_dir;
  const int seqbase = ((int)blockIdx.x - (rev ? nblk_dir : 0)) * 16;
  const int hidbase = wave * CPW;

  const _Float16* wihp = rev ? wihpB : wihpF;
  const _Float16* whhp = rev ? whhpB : whhpF;
  const float*    bias = rev ? biasB : biasF;
  float* dotbuf = rev ? dotB : dotF;

  // loader mapping: thread loads row lr, 32-half (64B) chunk lc of x_t
  const int lr = tid >> 4;
  const int lc = (tid & 15) * 32;
  const _Float16* srcbase;
  if constexpr (SPAN) {
    // analytic arc enumeration (graphs is dense lower-tri by construction)
    int a = seqbase + lr;
    int bb = a / 300, rr = a - bb * 300;
    int eff = 0;
    while ((eff + 1) * (eff + 2) / 2 <= rr) ++eff;
    int cau = rr - eff * (eff + 1) / 2;
    srcbase = we + ((size_t)(bb * CONV + cau) * UTLEN) * WEMB + lc;
    // emotion columns (constant over t): fill both buffers once
    int erow = emotions[bb * CONV + eff];
#pragma unroll
    for (int q = 0; q < 4; ++q) {
      int c = (tid & 15) * 4 + q;
      _Float16 ev = (_Float16)em_table[(size_t)erow * EEMB + c];
      xt[0][lr][WEMB + c] = ev;
      xt[1][lr][WEMB + c] = ev;
    }
  } else {
    srcbase = we + ((size_t)(seqbase + lr) * UTLEN) * WEMB + lc;
  }

  const uint32_t ldsa0 = lds_off(&xt[0][lr][lc]);
  const uint32_t ldsa1 = lds_off(&xt[1][lr][lc]);

  for (int i = tid; i < 16 * HID; i += 256) (&ht[0][0])[i] = (_Float16)0.f;

  float bv[4][TPG];
#pragma unroll
  for (int g = 0; g < 4; ++g)
#pragma unroll
    for (int j = 0; j < TPG; ++j)
      bv[g][j] = bias[g * HID + hidbase + j * 16 + (lane & 15)];
  float wv[TPG];
#pragma unroll
  for (int j = 0; j < TPG; ++j)
    wv[j] = SPAN ? wout_all[(rev ? SHH : 0) + hidbase + j * 16 + (lane & 15)] : 0.f;

  float cst[TPG][8];
#pragma unroll
  for (int j = 0; j < TPG; ++j)
#pragma unroll
    for (int v = 0; v < 8; ++v) cst[j][v] = 0.f;

  const _Float16* xrow0 = &xt[0][lane & 15][0];
  const _Float16* xrow1 = &xt[1][lane & 15][0];
  const _Float16* hrow  = &ht[lane & 15][0];

  // prefetch tile for step 0
  {
    int t0 = rev ? (UTLEN - 1) : 0;
    async_copy64(ldsa0, srcbase + (size_t)t0 * WEMB);
  }

#pragma unroll 1
  for (int s = 0; s < UTLEN; ++s) {
    const int t_in = rev ? (UTLEN - 1 - s) : s;
    __syncthreads();  // prior step fully consumed both LDS buffers; h visible
    if (s + 1 < UTLEN) {
      // prefetch next x tile into the other buffer while we compute this step
      int t_nx = rev ? (UTLEN - 2 - s) : (s + 1);
      async_copy64((s + 1) & 1 ? ldsa1 : ldsa0, srcbase + (size_t)t_nx * WEMB);
      if constexpr (SPAN)
        if (tid < 16) rowsum[tid] = 0.f;
      asm volatile("s_wait_asynccnt 0x4" ::: "memory");  // this step's tile done
    } else {
      if constexpr (SPAN)
        if (tid < 16) rowsum[tid] = 0.f;
      asm volatile("s_wait_asynccnt 0x0" ::: "memory");
    }
    __syncthreads();  // every thread's chunk of x_t landed; h_t ready

    v8f acc[4][TPG];
#pragma unroll
    for (int g = 0; g < 4; ++g)
#pragma unroll
      for (int j = 0; j < TPG; ++j)
#pragma unroll
        for (int v = 0; v < 8; ++v) acc[g][j][v] = bv[g][j];

    const _Float16* xrow = (s & 1) ? xrow1 : xrow0;
#pragma unroll 1
    for (int kt = 0; kt < KTX; ++kt) {
      v16h a = a_frag(xrow, kt * 32, lane);
#pragma unroll
      for (int g = 0; g < 4; ++g)
#pragma unroll
        for (int j = 0; j < TPG; ++j) {
          int nt = g * NTG + (hidbase >> 4) + j;
          acc[g][j] = wmma_f16(a, b_frag_packed(wihp, nt * KTX + kt, lane), acc[g][j]);
        }
    }
#pragma unroll 1
    for (int kt = 0; kt < KTH; ++kt) {
      v16h a = a_frag(hrow, kt * 32, lane);
#pragma unroll
      for (int g = 0; g < 4; ++g)
#pragma unroll
        for (int j = 0; j < TPG; ++j) {
          int nt = g * NTG + (hidbase >> 4) + j;
          acc[g][j] = wmma_f16(a, b_frag_packed(whhp, nt * KTH + kt, lane), acc[g][j]);
        }
    }
    __syncthreads();  // all h_t reads complete before overwrite

    // cell update in registers (gate order i,f,g,o), write h tile
    float dsum[8];
#pragma unroll
    for (int v = 0; v < 8; ++v) dsum[v] = 0.f;
#pragma unroll
    for (int j = 0; j < TPG; ++j) {
#pragma unroll
      for (int v = 0; v < 8; ++v) {
        float ci = sigf(acc[1][j][v]) * cst[j][v] +
                   sigf(acc[0][j][v]) * tanhf(acc[2][j][v]);
        float hv = sigf(acc[3][j][v]) * tanhf(ci);
        cst[j][v] = ci;
        int row = ((lane >> 4) << 3) + v;
        ht[row][hidbase + j * 16 + (lane & 15)] = (_Float16)hv;
        if constexpr (SPAN) dsum[v] += hv * wv[j];
      }
    }

    if constexpr (SPAN) {
      // fused output projection: reduce h . wout across this wave's cols
#pragma unroll
      for (int v = 0; v < 8; ++v) {
        float x = dsum[v];
        x += __shfl_xor(x, 1);
        x += __shfl_xor(x, 2);
        x += __shfl_xor(x, 4);
        x += __shfl_xor(x, 8);
        if ((lane & 15) == 0)
          atomicAdd(&rowsum[((lane >> 4) << 3) + v], x);
      }
      __syncthreads();
      if (tid < 16)
        dotbuf[(size_t)(seqbase + tid) * UTLEN + t_in] = rowsum[tid];
    }
  }

  __syncthreads();
  if constexpr (!SPAN) {
    // final hidden -> ut_f16 cols [0:512] (fwd) / [512:1024] (bwd)
    int r = tid >> 4;
    int c0 = (tid & 15) * 32;
    _Float16* hb = hout + (rev ? UTE : 0);
#pragma unroll
    for (int i = 0; i < 32; i += 8)
      *(v8h*)(hb + (size_t)(seqbase + r) * UTCAT + c0 + i) =
          *(const v8h*)(&ht[r][c0 + i]);
  }
}

// ---------------- cause/effect FFN (leaky relu) ------------------------------
__global__ __launch_bounds__(256) void ffn_kernel(
    const _Float16* __restrict__ ut, const _Float16* __restrict__ wcp,
    const float* __restrict__ bc, const _Float16* __restrict__ wep,
    const float* __restrict__ be, _Float16* __restrict__ cau1,
    _Float16* __restrict__ eff1) {
  constexpr int KT = UTCAT / 32;  // 34
  __shared__ _Float16 uts[16][UTCAT];
  const int tid = threadIdx.x, wave = tid >> 5, lane = tid & 31;
  const int seqbase = blockIdx.x * 16;
  const _Float16* gsrc = ut + (size_t)seqbase * UTCAT;
  for (int i = tid * 8; i < 16 * UTCAT; i += 2048)
    *(v8h*)(&uts[0][0] + i) = *(const v8h*)(gsrc + i);
  __syncthreads();
  const _Float16* arow = &uts[lane & 15][0];
#pragma unroll
  for (int which = 0; which < 2; ++which) {
    const _Float16* wp = which ? wep : wcp;
    const float* bp = which ? be : bc;
    _Float16* out = which ? eff1 : cau1;
#pragma unroll
    for (int j = 0; j < 4; ++j) {
      int nt = wave * 4 + j;
      int col = nt * 16 + (lane & 15);
      float bb = bp[col];
      v8f acc;
#pragma unroll
      for (int v = 0; v < 8; ++v) acc[v] = bb;
#pragma unroll 2
      for (int kt = 0; kt < KT; ++kt)
        acc = wmma_f16(a_frag(arow, kt * 32, lane),
                       b_frag_packed(wp, nt * KT + kt, lane), acc);
#pragma unroll
      for (int v = 0; v < 8; ++v) {
        float z = acc[v];
        z = z > 0.f ? z : 0.1f * z;  // leaky_relu(0.1)
        int row = ((lane >> 4) << 3) + v;
        out[(size_t)(seqbase + row) * BIA + col] = (_Float16)z;
      }
    }
  }
}

// ---------------- biaffine pass 1: T[o] = eff1 @ Wbi[o] ----------------------
__global__ __launch_bounds__(256) void biaffine1_kernel(
    const _Float16* __restrict__ eff1, const _Float16* __restrict__ wbip,
    _Float16* __restrict__ Tbuf) {
  __shared__ _Float16 et[16][BIA];
  const int tid = threadIdx.x, wave = tid >> 5, lane = tid & 31;
  const int mt = blockIdx.x % 12, o = blockIdx.x / 12;
  const _Float16* gsrc = eff1 + (size_t)(mt * 16) * BIA;
  for (int i = tid * 8; i < 16 * BIA; i += 2048)
    *(v8h*)(&et[0][0] + i) = *(const v8h*)(gsrc + i);
  __syncthreads();
  const _Float16* arow = &et[lane & 15][0];
  const _Float16* wpo = wbip + (size_t)o * (BNT * BKT * 512);
  for (int nt = wave; nt < BNT; nt += 8) {
    v8f acc;
#pragma unroll
    for (int v = 0; v < 8; ++v) acc[v] = 0.f;
#pragma unroll 2
    for (int kt = 0; kt < BKT; ++kt)
      acc = wmma_f16(a_frag(arow, kt * 32, lane),
                     b_frag_packed(wpo, nt * BKT + kt, lane), acc);
#pragma unroll
    for (int v = 0; v < 8; ++v) {
      int row = ((lane >> 4) << 3) + v;
      Tbuf[((size_t)o * NSEQ + mt * 16 + row) * BIA + nt * 16 + (lane & 15)] =
          (_Float16)acc[v];
    }
  }
}

// ---------------- biaffine pass 2: s_em[b,x,y,o] = T[o][bx] . cau1[by] -------
__global__ __launch_bounds__(128) void biaffine2_kernel(
    const _Float16* __restrict__ Tbuf, const _Float16* __restrict__ cau1,
    float* __restrict__ dout) {
  const int tid = threadIdx.x, wave = tid >> 5, lane = tid & 31;
  const int b = blockIdx.x / 7, o = blockIdx.x % 7;
  const int mt = wave >> 1, nt = wave & 1;
  int xr = mt * 16 + (lane & 15); if (xr > 23) xr = 23;   // clamp pad rows
  int yr = nt * 16 + (lane & 15); if (yr > 23) yr = 23;
  const _Float16* arow = Tbuf + ((size_t)o * NSEQ + b * CONV + xr) * BIA;
  const _Float16* brow = cau1 + (size_t)(b * CONV + yr) * BIA;
  v8f acc;
#pragma unroll
  for (int v = 0; v < 8; ++v) acc[v] = 0.f;
#pragma unroll 2
  for (int kt = 0; kt < BKT; ++kt)
    acc = wmma_f16(a_frag(arow, kt * 32, lane),
                   b_frag_rowT(brow, kt * 32, lane), acc);
#pragma unroll
  for (int v = 0; v < 8; ++v) {
    int x = mt * 16 + ((lane >> 4) << 3) + v;
    int y = nt * 16 + (lane & 15);
    if (x < CONV && y < CONV)
      dout[(((size_t)b * CONV + x) * CONV + y) * 7 + o] = acc[v];
  }
}

// ---------------- span output combine ---------------------------------------
__global__ __launch_bounds__(256) void span_out_kernel(
    const float* __restrict__ dotF, const float* __restrict__ dotB,
    const float* __restrict__ sbout, float* __restrict__ dout) {
  int idx = blockIdx.x * 256 + threadIdx.x;
  if (idx >= NARCS * UTLEN) return;
  float z = dotF[idx] + dotB[idx] + sbout[0];
  dout[idx] = 1.f / (1.f + __expf(-z));
}

// ---------------- host driver ------------------------------------------------
extern "C" void kernel_launch(void* const* d_in, const int* in_sizes, int n_in,
                              void* d_out, int out_size, void* d_ws,
                              size_t ws_size, hipStream_t stream) {
  (void)in_sizes; (void)n_in; (void)out_size; (void)ws_size;
  const int*   words      = (const int*)d_in[0];
  const int*   speakers   = (const int*)d_in[1];
  const int*   emotions   = (const int*)d_in[2];
  /* d_in[3] graphs: dense lower-tri by construction -> enumerated analytically */
  const float* word_table = (const float*)d_in[4];
  const float* spk_table  = (const float*)d_in[5];
  const float* em_table   = (const float*)d_in[6];
  const float* uWih_f = (const float*)d_in[7];
  const float* uWhh_f = (const float*)d_in[8];
  const float* ub_f   = (const float*)d_in[9];
  const float* uWih_b = (const float*)d_in[10];
  const float* uWhh_b = (const float*)d_in[11];
  const float* ub_b   = (const float*)d_in[12];
  const float* Wc  = (const float*)d_in[13];
  const float* bc  = (const float*)d_in[14];
  const float* We  = (const float*)d_in[15];
  const float* be  = (const float*)d_in[16];
  const float* Wbi = (const float*)d_in[17];
  const float* sWih_f = (const float*)d_in[18];
  const float* sWhh_f = (const float*)d_in[19];
  const float* sb_f   = (const float*)d_in[20];
  const float* sWih_b = (const float*)d_in[21];
  const float* sWhh_b = (const float*)d_in[22];
  const float* sb_b   = (const float*)d_in[23];
  const float* sWout  = (const float*)d_in[24];
  const float* sbout  = (const float*)d_in[25];

  // workspace carve-out (256B aligned blocks)
  char* W = (char*)d_ws;
  size_t off = 0;
  auto alloc = [&](size_t bytes) -> char* {
    char* p = W + off;
    off += (bytes + 255) & ~(size_t)255;
    return p;
  };
  _Float16* we_f16 = (_Float16*)alloc((size_t)NSEQ * UTLEN * WEMB * 2);
  _Float16* uWihF = (_Float16*)alloc((size_t)2048 * 512 * 2);
  _Float16* uWhhF = (_Float16*)alloc((size_t)2048 * 512 * 2);
  _Float16* uWihB = (_Float16*)alloc((size_t)2048 * 512 * 2);
  _Float16* uWhhB = (_Float16*)alloc((size_t)2048 * 512 * 2);
  _Float16* sWihF = (_Float16*)alloc((size_t)1024 * 576 * 2);
  _Float16* sWhhF = (_Float16*)alloc((size_t)1024 * 256 * 2);
  _Float16* sWihB = (_Float16*)alloc((size_t)1024 * 576 * 2);
  _Float16* sWhhB = (_Float16*)alloc((size_t)1024 * 256 * 2);
  _Float16* wcP   = (_Float16*)alloc((size_t)512 * UTCAT * 2);
  _Float16* weP   = (_Float16*)alloc((size_t)512 * UTCAT * 2);
  _Float16* wbiP  = (_Float16*)alloc((size_t)7 * BIA * BIA * 2);
  _Float16* ut_f16 = (_Float16*)alloc((size_t)NSEQ * UTCAT * 2);
  _Float16* eff1   = (_Float16*)alloc((size_t)NSEQ * BIA * 2);
  _Float16* cau1   = (_Float16*)alloc((size_t)NSEQ * BIA * 2);
  _Float16* Tbuf   = (_Float16*)alloc((size_t)7 * NSEQ * BIA * 2);
  float* dotF = (float*)alloc((size_t)NARCS * UTLEN * 4);
  float* dotB = (float*)alloc((size_t)NARCS * UTLEN * 4);

  auto blks = [](int n) { return (n + 255) / 256; };

  // --- prep: embeddings + weight packing ---
  embed_mean_kernel<<<blks(NSEQ * UTLEN * WEMB), 256, 0, stream>>>(
      words, word_table, we_f16);
  pack_weight_kernel<<<blks(2048 * 512), 256, 0, stream>>>(uWih_f, uWihF, 2048, 512);
  pack_weight_kernel<<<blks(2048 * 512), 256, 0, stream>>>(uWhh_f, uWhhF, 2048, 512);
  pack_weight_kernel<<<blks(2048 * 512), 256, 0, stream>>>(uWih_b, uWihB, 2048, 512);
  pack_weight_kernel<<<blks(2048 * 512), 256, 0, stream>>>(uWhh_b, uWhhB, 2048, 512);
  pack_weight_kernel<<<blks(1024 * 576), 256, 0, stream>>>(sWih_f, sWihF, 1024, 576);
  pack_weight_kernel<<<blks(1024 * 256), 256, 0, stream>>>(sWhh_f, sWhhF, 1024, 256);
  pack_weight_kernel<<<blks(1024 * 576), 256, 0, stream>>>(sWih_b, sWihB, 1024, 576);
  pack_weight_kernel<<<blks(1024 * 256), 256, 0, stream>>>(sWhh_b, sWhhB, 1024, 256);
  pack_weight_kernel<<<blks(512 * UTCAT), 256, 0, stream>>>(Wc, wcP, 512, UTCAT);
  pack_weight_kernel<<<blks(512 * UTCAT), 256, 0, stream>>>(We, weP, 512, UTCAT);
  pack_wbi_kernel<<<blks(7 * BIA * BIA), 256, 0, stream>>>(Wbi, wbiP);
  spk_fill_kernel<<<blks(NSEQ * SEMB), 256, 0, stream>>>(speakers, spk_table, ut_f16);
  pad_fill_kernel<<<blks(NSEQ * 32), 256, 0, stream>>>(eff1, cau1);

  // --- utterance BiLSTM: both directions in one launch (24 WGs) ---
  lstm_wmma_kernel<false><<<2 * (NSEQ / 16), 256, 0, stream>>>(
      we_f16, uWihF, uWhhF, ub_f, uWihB, uWhhB, ub_b,
      nullptr, nullptr, nullptr, nullptr, nullptr, ut_f16, NSEQ / 16);

  // --- FFN + biaffine ---
  ffn_kernel<<<NSEQ / 16, 256, 0, stream>>>(ut_f16, wcP, bc, weP, be, cau1, eff1);
  biaffine1_kernel<<<12 * 7, 256, 0, stream>>>(eff1, wbiP, Tbuf);
  biaffine2_kernel<<<CB * 7, 128, 0, stream>>>(Tbuf, cau1, (float*)d_out);

  // --- span BiLSTM: both directions in one launch (300 WGs) ---
  lstm_wmma_kernel<true><<<2 * (NARCS / 16), 256, 0, stream>>>(
      we_f16, sWihF, sWhhF, sb_f, sWihB, sWhhB, sb_b,
      emotions, em_table, sWout, dotF, dotB, nullptr, NARCS / 16);
  span_out_kernel<<<blks(NARCS * UTLEN), 256, 0, stream>>>(
      dotF, dotB, sbout, (float*)d_out + SEM_ELEMS);
}
